// HierarchicalGRUDecoder_11957188952341
// MI455X (gfx1250) — matrix operand order repaired and checked by reading, not compiled, not run on hardware
//
#include <hip/hip_runtime.h>
#include <cstdint>
#include <cstddef>

#define NE_ 16
#define S_  16
#define B_  256
#define IN_ 61
#define INP 64    // padded K for the x-part GEMM
#define H_  1024
#define L_  512
#define NL_ 2
#define H3  (3 * H_)

typedef __attribute__((ext_vector_type(16))) __bf16 v16bf;
typedef __attribute__((ext_vector_type(8)))  __bf16 v8bf;
typedef __attribute__((ext_vector_type(8)))  float  v8f;

#define TM   128
#define TN   128
#define TK   32
#define LDSK 40   // padded LDS row stride in bf16 elems (80B): 16B-aligned rows

union FragU { v16bf v; v8bf h[2]; };
union AccU  { v8f  v; float f[8]; };

// A fragment (16x32 bf16): lane%16 = M; lanes 0-15: K{0-7,16-23}, lanes 16-31: K{8-15,24-31}
__device__ __forceinline__ v16bf fragA(const __bf16* tile, int row, int hf) {
  const __bf16* p = tile + row * LDSK;
  FragU u;
  u.h[0] = *(const v8bf*)(p + hf * 8);
  u.h[1] = *(const v8bf*)(p + 16 + hf * 8);
  return u.v;
}
// B fragment (32x16 bf16): lane%16 = N; lanes 0-15: K{0-7,16-23}, lanes 16-31: K{8-15,24-31}
__device__ __forceinline__ v16bf fragB(const __bf16* tile, int row, int hf) {
  const __bf16* p = tile + row * LDSK;
  FragU u;
  u.h[0] = *(const v8bf*)(p + hf * 8);
  u.h[1] = *(const v8bf*)(p + 16 + hf * 8);
  return u.v;
}

// C[M,N] = A[M,K](bf16) @ W[N,K]^T(bf16) + bias[N].  Requirements (guaranteed by
// host-side padding): M % 128 == 0, K % 32 == 0, W has ceil(N/128)*128 rows
// allocated. Only the epilogue guards gn < N. act=1 -> tanh. Cb: optional bf16
// copy of C.
__global__ __launch_bounds__(256) void gemm_bf16_nt(
    const __bf16* __restrict__ A, const __bf16* __restrict__ W,
    const float* __restrict__ bias, float* __restrict__ C,
    __bf16* __restrict__ Cb, int M, int N, int K, int act)
{
  __shared__ __align__(16) __bf16 sA[TM * LDSK];
  __shared__ __align__(16) __bf16 sB[TN * LDSK];
  const int tid   = threadIdx.x;
  const int m_blk = blockIdx.y * TM;
  const int n_blk = blockIdx.x * TN;
  const int wave  = tid >> 5;
  const int lane  = tid & 31;
  const int hf    = lane >> 4;
  const int lr    = lane & 15;
  const int wm    = (wave >> 1) * 32;
  const int wn    = (wave & 1) * 64;

  // staging assignment: 64 rows x 4 16B-chunks per pass, 2 passes per matrix
  const int rr = tid >> 2;          // 0..63
  const int cc = (tid & 3) * 8;     // 0,8,16,24 (bf16 elems)
  const __bf16* pA0 = A + (size_t)(m_blk + rr) * K + cc;
  const __bf16* pA1 = pA0 + (size_t)64 * K;
  const __bf16* pW0 = W + (size_t)(n_blk + rr) * K + cc;
  const __bf16* pW1 = pW0 + (size_t)64 * K;

  v8bf ra0 = *(const v8bf*)pA0;
  v8bf ra1 = *(const v8bf*)pA1;
  v8bf rb0 = *(const v8bf*)pW0;
  v8bf rb1 = *(const v8bf*)pW1;

  v8f acc[2][4] = {};

  for (int k0 = 0; k0 < K; k0 += TK) {
    *(v8bf*)(&sA[rr * LDSK + cc])        = ra0;
    *(v8bf*)(&sA[(rr + 64) * LDSK + cc]) = ra1;
    *(v8bf*)(&sB[rr * LDSK + cc])        = rb0;
    *(v8bf*)(&sB[(rr + 64) * LDSK + cc]) = rb1;
    __syncthreads();

    if (k0 + TK < K) {                 // prefetch next K-tile (overlaps WMMA)
      pA0 += TK; pA1 += TK; pW0 += TK; pW1 += TK;
      ra0 = *(const v8bf*)pA0;
      ra1 = *(const v8bf*)pA1;
      rb0 = *(const v8bf*)pW0;
      rb1 = *(const v8bf*)pW1;
    }

    v16bf a0 = fragA(sA, wm + lr, hf);
    v16bf a1 = fragA(sA, wm + 16 + lr, hf);
    v16bf b0 = fragB(sB, wn + lr, hf);
    v16bf b1 = fragB(sB, wn + 16 + lr, hf);
    v16bf b2 = fragB(sB, wn + 32 + lr, hf);
    v16bf b3 = fragB(sB, wn + 48 + lr, hf);

    acc[0][0] = __builtin_amdgcn_wmma_f32_16x16x32_bf16(false, a0, false, b0, (short)0, acc[0][0], false, false);
    acc[0][1] = __builtin_amdgcn_wmma_f32_16x16x32_bf16(false, a0, false, b1, (short)0, acc[0][1], false, false);
    acc[0][2] = __builtin_amdgcn_wmma_f32_16x16x32_bf16(false, a0, false, b2, (short)0, acc[0][2], false, false);
    acc[0][3] = __builtin_amdgcn_wmma_f32_16x16x32_bf16(false, a0, false, b3, (short)0, acc[0][3], false, false);
    acc[1][0] = __builtin_amdgcn_wmma_f32_16x16x32_bf16(false, a1, false, b0, (short)0, acc[1][0], false, false);
    acc[1][1] = __builtin_amdgcn_wmma_f32_16x16x32_bf16(false, a1, false, b1, (short)0, acc[1][1], false, false);
    acc[1][2] = __builtin_amdgcn_wmma_f32_16x16x32_bf16(false, a1, false, b2, (short)0, acc[1][2], false, false);
    acc[1][3] = __builtin_amdgcn_wmma_f32_16x16x32_bf16(false, a1, false, b3, (short)0, acc[1][3], false, false);
    __syncthreads();
  }

#pragma unroll
  for (int i = 0; i < 2; ++i) {
#pragma unroll
    for (int j = 0; j < 4; ++j) {
      AccU u; u.v = acc[i][j];
      int gn = n_blk + wn + j * 16 + lr;
      if (gn < N) {
        float bv = bias ? bias[gn] : 0.f;
#pragma unroll
        for (int v = 0; v < 8; ++v) {
          int gm = m_blk + wm + i * 16 + hf * 8 + v;   // D: VGPR v -> M = v + 8*half
          float val = u.f[v] + bv;
          if (act) val = tanhf(val);
          C[(size_t)gm * N + gn] = val;
          if (Cb) Cb[(size_t)gm * N + gn] = (__bf16)val;
        }
      }
    }
  }
}

// dst[r, c] (dst_rows x dst_cols, bf16) = src[r, col0 + c] for r < src_rows &&
// c < ncols_src, else 0. Used for weight conversion with row/col zero-padding.
__global__ void cvt_pad_bf16(const float* __restrict__ src, __bf16* __restrict__ dst,
                             int src_rows, int src_cols, int col0, int ncols_src,
                             int dst_rows, int dst_cols)
{
  int idx = blockIdx.x * blockDim.x + threadIdx.x;
  if (idx >= dst_rows * dst_cols) return;
  int r = idx / dst_cols, c = idx - r * dst_cols;
  float v = (r < src_rows && c < ncols_src) ? src[(size_t)r * src_cols + col0 + c] : 0.f;
  dst[idx] = (__bf16)v;
}

// xre_bf[s, ne, b, c] (c<INP, zero-padded past IN_) = target[ne*S + s, b, c]
__global__ void reorder_x_bf(const float* __restrict__ target, __bf16* __restrict__ xre)
{
  int idx = blockIdx.x * blockDim.x + threadIdx.x;
  const int total = S_ * NE_ * B_ * INP;
  if (idx >= total) return;
  int c  = idx % INP;
  int m  = idx / INP;
  int b  = m % B_;
  int ne = (m / B_) % NE_;
  int s  = m / (B_ * NE_);
  float v = (c < IN_) ? target[((size_t)(ne * S_ + s) * B_ + b) * IN_ + c] : 0.f;
  xre[idx] = (__bf16)v;
}

// hd[ne*B+b, j] = h0dec[ne, layer, b, j]  (fp32 carry + bf16 GEMM copy)
__global__ void reorder_h0dec(const float* __restrict__ h0dec, float* __restrict__ hd,
                              __bf16* __restrict__ hdb, int layer)
{
  int idx = blockIdx.x * blockDim.x + threadIdx.x;
  const int total = NE_ * B_ * H_;
  if (idx >= total) return;
  int j  = idx % H_;
  int b  = (idx / H_) % B_;
  int ne = idx / (H_ * B_);
  float v = h0dec[(((size_t)ne * NL_ + layer) * B_ + b) * H_ + j];
  hd[idx]  = v;
  hdb[idx] = (__bf16)v;
}

// PyTorch GRUCell combine; gi/gh [M,3H] with biases already added by the GEMMs;
// gi2 optional extra gi contribution. Writes fp32 carry + bf16 copy for GEMMs.
__global__ void gru_combine(const float* __restrict__ gi, const float* __restrict__ gi2,
                            const float* __restrict__ gh, const float* __restrict__ hprev,
                            float* __restrict__ hout, __bf16* __restrict__ hout_bf, int M)
{
  int idx = blockIdx.x * blockDim.x + threadIdx.x;
  if (idx >= M * H_) return;
  int b = idx / H_, j = idx - b * H_;
  size_t base = (size_t)b * H3;
  float ir = gi[base + j], iz = gi[base + H_ + j], inn = gi[base + 2 * H_ + j];
  if (gi2) {
    ir  += gi2[base + j];
    iz  += gi2[base + H_ + j];
    inn += gi2[base + 2 * H_ + j];
  }
  float hr = gh[base + j], hz = gh[base + H_ + j], hn = gh[base + 2 * H_ + j];
  float r = 1.f / (1.f + __expf(-(ir + hr)));
  float z = 1.f / (1.f + __expf(-(iz + hz)));
  float n = tanhf(inn + r * hn);
  float h = (1.f - z) * n + z * hprev[idx];
  hout[idx]    = h;
  hout_bf[idx] = (__bf16)h;
}

// one wave32 per row of [NE*B, IN_] logits; scatter into out[(ne*S+s), b, :]
__global__ void softmax_scatter(const float* __restrict__ logits, float* __restrict__ out, int s)
{
  int wv   = threadIdx.x >> 5;
  int lane = threadIdx.x & 31;
  int row  = blockIdx.x * 8 + wv;
  if (row >= NE_ * B_) return;
  const float* lp = logits + (size_t)row * IN_;
  float mx = -3.4e38f;
  for (int i = lane; i < IN_; i += 32) mx = fmaxf(mx, lp[i]);
  for (int o = 16; o > 0; o >>= 1) mx = fmaxf(mx, __shfl_xor(mx, o, 32));
  float sum = 0.f;
  for (int i = lane; i < IN_; i += 32) sum += __expf(lp[i] - mx);
  for (int o = 16; o > 0; o >>= 1) sum += __shfl_xor(sum, o, 32);
  float inv = 1.f / sum;
  int ne = row / B_, b = row - ne * B_;
  float* op = out + ((size_t)(ne * S_ + s) * B_ + b) * IN_;
  for (int i = lane; i < IN_; i += 32) op[i] = __expf(lp[i] - mx) * inv;
}

extern "C" void kernel_launch(void* const* d_in, const int* in_sizes, int n_in,
                              void* d_out, int out_size, void* d_ws, size_t ws_size,
                              hipStream_t stream)
{
  (void)in_sizes; (void)n_in; (void)out_size; (void)ws_size;
  const float* target = (const float*)d_in[0];
  const float* latent = (const float*)d_in[1];
  const float* h0     = (const float*)d_in[2];
  const float* h0dec  = (const float*)d_in[3];
  const float* c_wih0 = (const float*)d_in[4];
  const float* c_whh0 = (const float*)d_in[5];
  const float* c_bih0 = (const float*)d_in[6];
  const float* c_bhh0 = (const float*)d_in[7];
  const float* c_wih1 = (const float*)d_in[8];
  const float* c_whh1 = (const float*)d_in[9];
  const float* c_bih1 = (const float*)d_in[10];
  const float* c_bhh1 = (const float*)d_in[11];
  const float* ce_w   = (const float*)d_in[12];
  const float* ce_b   = (const float*)d_in[13];
  const float* d_wih0 = (const float*)d_in[14];
  const float* d_whh0 = (const float*)d_in[15];
  const float* d_bih0 = (const float*)d_in[16];
  const float* d_bhh0 = (const float*)d_in[17];
  const float* d_wih1 = (const float*)d_in[18];
  const float* d_whh1 = (const float*)d_in[19];
  const float* d_bih1 = (const float*)d_in[20];
  const float* d_bhh1 = (const float*)d_in[21];
  const float* o_w    = (const float*)d_in[22];
  const float* o_b    = (const float*)d_in[23];
  float* out = (float*)d_out;

  char* wp = (char*)d_ws;
  auto alloc = [&](size_t elems, size_t esz) -> void* {
    void* q = (void*)wp;
    wp += (elems * esz + 255) & ~(size_t)255;
    return q;
  };

  const int MB = NE_ * B_;   // 4096 decoder batch

  // ---- bf16 weights (o_w row-padded to 128, d_wx col-padded to 64) ----
  __bf16* wb_c_wih0 = (__bf16*)alloc((size_t)H3 * L_, 2);
  __bf16* wb_c_whh0 = (__bf16*)alloc((size_t)H3 * H_, 2);
  __bf16* wb_c_wih1 = (__bf16*)alloc((size_t)H3 * H_, 2);
  __bf16* wb_c_whh1 = (__bf16*)alloc((size_t)H3 * H_, 2);
  __bf16* wb_ce_w   = (__bf16*)alloc((size_t)L_ * H_, 2);
  __bf16* wb_d_wx   = (__bf16*)alloc((size_t)H3 * INP, 2);
  __bf16* wb_d_we   = (__bf16*)alloc((size_t)H3 * L_, 2);
  __bf16* wb_d_whh0 = (__bf16*)alloc((size_t)H3 * H_, 2);
  __bf16* wb_d_wih1 = (__bf16*)alloc((size_t)H3 * H_, 2);
  __bf16* wb_d_whh1 = (__bf16*)alloc((size_t)H3 * H_, 2);
  __bf16* wb_o_w    = (__bf16*)alloc((size_t)128 * H_, 2);

  // ---- bf16 activations (GEMM A operands) ----
  __bf16* latent_bf = (__bf16*)alloc((size_t)B_ * L_, 2);
  __bf16* h0l0_bf   = (__bf16*)alloc((size_t)B_ * H_, 2);
  __bf16* h0l1_bf   = (__bf16*)alloc((size_t)B_ * H_, 2);
  __bf16* h_c0_bf   = (__bf16*)alloc((size_t)B_ * H_, 2);
  __bf16* c_out_bf  = (__bf16*)alloc((size_t)NE_ * B_ * H_, 2);
  __bf16* emb_bf    = (__bf16*)alloc((size_t)MB * L_, 2);
  __bf16* xre_bf    = (__bf16*)alloc((size_t)S_ * MB * INP, 2);
  __bf16* h_d0_bf   = (__bf16*)alloc((size_t)MB * H_, 2);
  __bf16* h_d1_bf   = (__bf16*)alloc((size_t)MB * H_, 2);

  // ---- fp32 activations / scratch ----
  float* gi0c   = (float*)alloc((size_t)B_ * H3, 4);
  float* gibuf  = (float*)alloc((size_t)MB * H3, 4);
  float* ghbuf  = (float*)alloc((size_t)MB * H3, 4);
  float* h_c0   = (float*)alloc((size_t)B_ * H_, 4);
  float* c_out  = (float*)alloc((size_t)NE_ * B_ * H_, 4);
  float* emb    = (float*)alloc((size_t)MB * L_, 4);
  float* gi_emb = (float*)alloc((size_t)MB * H3, 4);
  float* h_d0   = (float*)alloc((size_t)MB * H_, 4);
  float* h_d1   = (float*)alloc((size_t)MB * H_, 4);
  float* logits = (float*)alloc((size_t)MB * IN_, 4);

  auto cvt = [&](const float* src, __bf16* dst, int sr, int sc, int c0, int nc,
                 int dr, int dc) {
    int total = dr * dc;
    cvt_pad_bf16<<<(total + 255) / 256, 256, 0, stream>>>(src, dst, sr, sc, c0, nc, dr, dc);
  };
  auto gemm = [&](const __bf16* A, const __bf16* W, const float* bias, float* C,
                  __bf16* Cb, int M, int N, int K, int act) {
    dim3 g((N + TN - 1) / TN, M / TM);
    gemm_bf16_nt<<<g, 256, 0, stream>>>(A, W, bias, C, Cb, M, N, K, act);
  };
  auto comb = [&](const float* gi, const float* gi2, const float* gh,
                  const float* hp, float* ho, __bf16* hob, int M) {
    gru_combine<<<((M * H_) + 255) / 256, 256, 0, stream>>>(gi, gi2, gh, hp, ho, hob, M);
  };

  // ---- setup: weight conversion + activation conversion / reorders ----
  cvt(c_wih0, wb_c_wih0, H3, L_, 0, L_, H3, L_);
  cvt(c_whh0, wb_c_whh0, H3, H_, 0, H_, H3, H_);
  cvt(c_wih1, wb_c_wih1, H3, H_, 0, H_, H3, H_);
  cvt(c_whh1, wb_c_whh1, H3, H_, 0, H_, H3, H_);
  cvt(ce_w,   wb_ce_w,   L_,  H_, 0, H_, L_,  H_);
  cvt(d_wih0, wb_d_wx,   H3, IN_ + L_, 0,   IN_, H3, INP);   // col pad 61->64
  cvt(d_wih0, wb_d_we,   H3, IN_ + L_, IN_, L_,  H3, L_);
  cvt(d_whh0, wb_d_whh0, H3, H_, 0, H_, H3, H_);
  cvt(d_wih1, wb_d_wih1, H3, H_, 0, H_, H3, H_);
  cvt(d_whh1, wb_d_whh1, H3, H_, 0, H_, H3, H_);
  cvt(o_w,    wb_o_w,    IN_, H_, 0, H_, 128, H_);           // row pad 61->128

  cvt(latent, latent_bf, B_, L_, 0, L_, B_, L_);
  cvt(h0,                  h0l0_bf, B_, H_, 0, H_, B_, H_);
  cvt(h0 + (size_t)B_ * H_, h0l1_bf, B_, H_, 0, H_, B_, H_);

  {
    int t1 = S_ * MB * INP;
    reorder_x_bf<<<(t1 + 255) / 256, 256, 0, stream>>>(target, xre_bf);
    int t2 = NE_ * B_ * H_;
    reorder_h0dec<<<(t2 + 255) / 256, 256, 0, stream>>>(h0dec, h_d0, h_d0_bf, 0);
    reorder_h0dec<<<(t2 + 255) / 256, 256, 0, stream>>>(h0dec, h_d1, h_d1_bf, 1);
  }

  // ---- conductor: layer-0 input GEMM is step-invariant ----
  gemm(latent_bf, wb_c_wih0, c_bih0, gi0c, nullptr, B_, H3, L_, 0);
  const float*  hc0_prev   = h0;                         // fp32 carry (zeros)
  const __bf16* hc0_prevb  = h0l0_bf;
  const float*  hc1_prev   = h0 + (size_t)B_ * H_;
  const __bf16* hc1_prevb  = h0l1_bf;
  for (int t = 0; t < NE_; ++t) {
    gemm(hc0_prevb, wb_c_whh0, c_bhh0, ghbuf, nullptr, B_, H3, H_, 0);
    comb(gi0c, nullptr, ghbuf, hc0_prev, h_c0, h_c0_bf, B_);
    hc0_prev = h_c0; hc0_prevb = h_c0_bf;
    gemm(h_c0_bf, wb_c_wih1, c_bih1, gibuf, nullptr, B_, H3, H_, 0);
    gemm(hc1_prevb, wb_c_whh1, c_bhh1, ghbuf, nullptr, B_, H3, H_, 0);
    float*  ho  = c_out    + (size_t)t * B_ * H_;        // h1 chain lives in c_out
    __bf16* hob = c_out_bf + (size_t)t * B_ * H_;
    comb(gibuf, nullptr, ghbuf, hc1_prev, ho, hob, B_);
    hc1_prev = ho; hc1_prevb = hob;
  }

  // emb = tanh(c_out @ ce_w^T + ce_b); bf16 copy feeds the next GEMM
  gemm(c_out_bf, wb_ce_w, ce_b, emb, emb_bf, MB, L_, H_, 1);
  // gi_emb = emb @ W_e^T (timestep-invariant part of decoder layer-0 input GEMM)
  gemm(emb_bf, wb_d_we, nullptr, gi_emb, nullptr, MB, H3, L_, 0);

  // ---- decoder (teacher forcing): 16 sequential steps at batch 4096 ----
  for (int s = 0; s < S_; ++s) {
    const __bf16* xs = xre_bf + (size_t)s * MB * INP;
    gemm(xs, wb_d_wx, d_bih0, gibuf, nullptr, MB, H3, INP, 0);   // x part + bih0
    gemm(h_d0_bf, wb_d_whh0, d_bhh0, ghbuf, nullptr, MB, H3, H_, 0);
    comb(gibuf, gi_emb, ghbuf, h_d0, h_d0, h_d0_bf, MB);
    gemm(h_d0_bf, wb_d_wih1, d_bih1, gibuf, nullptr, MB, H3, H_, 0);
    gemm(h_d1_bf, wb_d_whh1, d_bhh1, ghbuf, nullptr, MB, H3, H_, 0);
    comb(gibuf, nullptr, ghbuf, h_d1, h_d1, h_d1_bf, MB);
    gemm(h_d1_bf, wb_o_w, o_b, logits, nullptr, MB, IN_, H_, 0); // output head
    softmax_scatter<<<(MB + 7) / 8, 256, 0, stream>>>(logits, out, s);
  }
}
